// STGCN_52664888983571
// MI455X (gfx1250) — matrix operand order repaired and checked by reading, not compiled, not run on hardware
//
#include <hip/hip_runtime.h>
#include <hip/hip_bf16.h>
#include <math.h>

// ---------------- CDNA5 WMMA types ----------------
typedef __attribute__((ext_vector_type(16))) __bf16 v16bf;
typedef __attribute__((ext_vector_type(8)))  float  v8f;

#if __has_builtin(__builtin_amdgcn_global_load_async_to_lds_b128)
#define HAVE_ASYNC_LDS 1
#endif

#if __has_builtin(__builtin_amdgcn_s_wait_asynccnt)
#define ASYNC_WAIT(n) __builtin_amdgcn_s_wait_asynccnt(n)
#else
#define ASYNC_WAIT(n) asm volatile("s_wait_asynccnt %0" ::"i"(n))
#endif

#define WPITCH 72   // LDS weight row pitch (bf16): 36*col mod 64 banks all-distinct

static __device__ inline v8f wmma_bf16(v16bf a, v16bf b, v8f c) {
  // D = A(16x32 bf16) * B(32x16 bf16) + C(16x16 f32)
  return __builtin_amdgcn_wmma_f32_16x16x32_bf16(false, a, false, b, (short)0, c, false, false);
}

// Build A-tile (16x32, bf16) from f32 row-major [16 x ld], K-offset kt.
// ISA 16-bit A layout: lanes0-15 row=lane, els0-7 = K kt+kh..+7 (kh=0),
// els8-15 = K kt+16+kh..; lanes16-31 same rows with kh=8.
static __device__ inline v16bf make_a_tile_f32(const float* __restrict__ base, int ld,
                                               int kt, int lane) {
  int row = lane & 15;
  int kh  = (lane >> 4) * 8;
  const float* p0 = base + (size_t)row * ld + kt + kh;
  const float* p1 = p0 + 16;
  v16bf a;
#pragma unroll
  for (int i = 0; i < 8; ++i) { a[i] = (__bf16)p0[i]; a[i + 8] = (__bf16)p1[i]; }
  return a;
}

// Build B-tile (32x16, bf16) from Wt stored [N][K] bf16 with row pitch ldk.
// ISA B layout: col=lane&15; els0-7 = K kt+kh..+7, els8-15 = K kt+16+kh..+7, kh=8*(lane>=16).
static __device__ inline v16bf make_b_tile_bf16(const __bf16* Wt, int ldk,
                                                int nt, int kt, int lane) {
  const __bf16* p = Wt + (size_t)(nt * 16 + (lane & 15)) * ldk + kt + ((lane >> 4) * 8);
  v16bf b;
#pragma unroll
  for (int i = 0; i < 8; ++i) { b[i] = p[i]; b[i + 8] = p[i + 16]; }
  return b;
}

// Store C tile (16x16 f32). VGPR r -> row r + 8*(lane>=16), col = lane&15.
static __device__ inline void store_c_f32(float* __restrict__ base, int ld, int nt,
                                          int lane, v8f c) {
  int col = nt * 16 + (lane & 15);
  int r0  = (lane >> 4) * 8;
#pragma unroll
  for (int r = 0; r < 8; ++r) base[(size_t)(r0 + r) * ld + col] = c[r];
}

static __device__ inline float sigmoidf_(float x) { return 1.0f / (1.0f + __expf(-x)); }

#if defined(HAVE_ASYNC_LDS)
// b128 async payload type: pointer-to-v4i in global (AS1) / LDS (AS3) address spaces.
typedef int v4i_ __attribute__((__vector_size__(16)));
typedef __attribute__((address_space(1))) v4i_ gv4i;
typedef __attribute__((address_space(3))) v4i_ lv4i;

// Async-copy a contiguous 4KB tile (1024 f32) global -> LDS; 8 x b128 per wave.
static __device__ inline void async_copy_tile_4k(const float* __restrict__ g,
                                                 float* l, int lane) {
#pragma unroll
  for (int i = 0; i < 8; ++i) {
    __builtin_amdgcn_global_load_async_to_lds_b128(
        (gv4i*)(const_cast<float*>(g) + i * 128 + lane * 4),
        (lv4i*)(l + i * 128 + lane * 4),
        0, 0);
  }
}
#endif

// ---------------- graph norm ----------------
__global__ void zero_kernel(float* p, int n) {
  int i = blockIdx.x * blockDim.x + threadIdx.x;
  if (i < n) p[i] = 0.0f;
}

__global__ void deg_kernel(const long long* __restrict__ ei, float* __restrict__ deg, int E) {
  int e = blockIdx.x * blockDim.x + threadIdx.x;
  if (e < E) atomicAdd(&deg[(int)ei[(size_t)E + e]], 1.0f);
}

__global__ void dinv_kernel(float* __restrict__ deg, int n) {
  int i = blockIdx.x * blockDim.x + threadIdx.x;
  if (i < n) deg[i] = rsqrtf(deg[i] + 1.0f);   // +1 = self loop
}

// ---------------- weight prep: transpose + f32->bf16 ----------------
__global__ void prep_weights_kernel(const float* __restrict__ W1, const float* __restrict__ W2,
                                    const float* __restrict__ Wih, const float* __restrict__ Whh,
                                    __bf16* __restrict__ wt1, __bf16* __restrict__ wt2,
                                    __bf16* __restrict__ wihb, __bf16* __restrict__ whhb) {
  int t = blockIdx.x * blockDim.x + threadIdx.x;
  int stride = gridDim.x * blockDim.x;
  // Wt1: [64][32] (K padded 16->32 with zeros), Wt1[n][k] = W1[k][n]
  for (int i = t; i < 64 * 32; i += stride) {
    int n = i >> 5, k = i & 31;
    wt1[i] = (__bf16)((k < 16) ? W1[k * 64 + n] : 0.0f);
  }
  // Wt2: [64][64], Wt2[n][k] = W2[k][n]
  for (int i = t; i < 64 * 64; i += stride) {
    int n = i >> 6, k = i & 63;
    wt2[i] = (__bf16)W2[k * 64 + n];
  }
  // W_ih/W_hh already [256 out][64 in] == [N][K]
  for (int i = t; i < 256 * 64; i += stride) {
    wihb[i] = (__bf16)Wih[i];
    whhb[i] = (__bf16)Whh[i];
  }
}

// ---------------- GEMM (rows x K) @ Wt[64][K] -> (rows x 64), WMMA bf16 ----------------
template <int K, int LDW>
__global__ void gemm_kernel(const float* __restrict__ A, const __bf16* __restrict__ Wt,
                            float* __restrict__ C, int nrow_tiles) {
  int tile = blockIdx.x * (blockDim.x >> 5) + (threadIdx.x >> 5);
  if (tile >= nrow_tiles) return;           // wave-uniform -> EXEC all-1 where active
  int lane = threadIdx.x & 31;
  const float* ab = A + (size_t)tile * 16 * K;

  v16bf a0{}, a1{};
  if constexpr (K == 16) {  // pad K 16 -> 32 with zeros
    int row = lane & 15, kh = (lane >> 4) * 8;
    const float* p = ab + (size_t)row * 16 + kh;
#pragma unroll
    for (int i = 0; i < 8; ++i) { a0[i] = (__bf16)p[i]; a0[i + 8] = (__bf16)0.0f; }
  } else {                  // K == 64
    a0 = make_a_tile_f32(ab, 64, 0, lane);
    a1 = make_a_tile_f32(ab, 64, 32, lane);
  }

  float* cb = C + (size_t)tile * 16 * 64;
#pragma unroll
  for (int nt = 0; nt < 4; ++nt) {
    v8f c = {};
    c = wmma_bf16(a0, make_b_tile_bf16(Wt, LDW, nt, 0, lane), c);
    if constexpr (K == 64) {
      c = wmma_bf16(a1, make_b_tile_bf16(Wt, LDW, nt, 32, lane), c);
    }
    store_c_f32(cb, 64, nt, lane, c);
  }
}

// ---------------- aggregation: init with self-loop term ----------------
__global__ void agg_init_kernel(const float* __restrict__ h, const float* __restrict__ dinv,
                                float* __restrict__ agg, int N) {
  size_t idx = (size_t)blockIdx.x * blockDim.x + threadIdx.x;   // float2 index
  size_t total = (size_t)256 * N * 32;
  if (idx >= total) return;
  int n = (int)((idx >> 5) % (size_t)N);
  float w = dinv[n]; w = w * w;
  float2 v = ((const float2*)h)[idx];
  float2 o; o.x = w * v.x; o.y = w * v.y;
  ((float2*)agg)[idx] = o;
}

// one wave per edge, grid.y = bt slice (keeps the 2.56MB slice L2-resident)
__global__ void agg_scatter_kernel(const float* __restrict__ h, const float* __restrict__ dinv,
                                   const long long* __restrict__ ei, float* __restrict__ agg,
                                   int E, int N) {
  int e = blockIdx.x * (blockDim.x >> 5) + (threadIdx.x >> 5);
  if (e >= E) return;
  int bt   = blockIdx.y;
  int lane = threadIdx.x & 31;
  int s = (int)ei[e];
  int d = (int)ei[(size_t)E + e];
  float w = dinv[s] * dinv[d];
  const float2* hp = (const float2*)(h + ((size_t)bt * N + s) * 64);
  float*        ap = agg + ((size_t)bt * N + d) * 64;
  float2 v = hp[lane];
  atomicAdd(ap + 2 * lane,     w * v.x);
  atomicAdd(ap + 2 * lane + 1, w * v.y);
}

__global__ void bias_act_kernel(float* __restrict__ agg, const float* __restrict__ b,
                                int relu, size_t total) {
  size_t idx = (size_t)blockIdx.x * blockDim.x + threadIdx.x;
  if (idx >= total) return;
  float v = agg[idx] + b[idx & 63];
  if (relu) v = fmaxf(v, 0.0f);
  agg[idx] = v;
}

// ---------------- fused LSTM over T=32 + final linear ----------------
// x layout: [B,T,N,64] f32 (aggbuf). Sequences s = b*N + n; 16 consecutive n per wave,
// so each wave's per-timestep x tile is one contiguous 4KB block -> async-to-LDS friendly.
// Weights live in LDS (pitch-72 padded, bank-conflict-free B-tiles).
__global__ void lstm_kernel(const float* __restrict__ xin,
                            const __bf16* __restrict__ wih, const __bf16* __restrict__ whh,
                            const float* __restrict__ bih, const float* __restrict__ bhh,
                            const float* __restrict__ wlin, const float* __restrict__ blin,
                            float* __restrict__ out, int N, int T) {
  __shared__ __bf16 swih[256 * WPITCH];     // 36KB
  __shared__ __bf16 swhh[256 * WPITCH];     // 36KB
  __shared__ __bf16 shh[8][16 * 64];        // per-wave h state, bf16 (16KB)
#if defined(HAVE_ASYNC_LDS)
  __shared__ float  sx[8][2][16 * 64];      // per-wave x double buffer (64KB)
#endif
  int wv = threadIdx.x >> 5, lane = threadIdx.x & 31;
  int s0 = blockIdx.x * 128 + wv * 16;      // 80000 = 625 blocks * 128 exact, 16 | N
  int b  = s0 / N, n0 = s0 % N;
  __bf16* hbuf = shh[wv];

  // --- stage weights into LDS (pitch WPITCH) ---
#if defined(HAVE_ASYNC_LDS)
#pragma unroll
  for (int i = 0; i < 8; ++i) {             // 2048 16B-chunks over 8 waves
    int id  = wv * 256 + i * 32 + lane;
    int row = id >> 3, cc = (id & 7) * 8;   // row 0..255, col offset in bf16
    __builtin_amdgcn_global_load_async_to_lds_b128(
        (gv4i*)(const_cast<__bf16*>(wih) + row * 64 + cc),
        (lv4i*)(swih + row * WPITCH + cc), 0, 0);
    __builtin_amdgcn_global_load_async_to_lds_b128(
        (gv4i*)(const_cast<__bf16*>(whh) + row * 64 + cc),
        (lv4i*)(swhh + row * WPITCH + cc), 0, 0);
  }
#else
  for (int i = threadIdx.x; i < 256 * 64; i += 256) {
    int row = i >> 6, k = i & 63;
    swih[row * WPITCH + k] = wih[i];
    swhh[row * WPITCH + k] = whh[i];
  }
#endif

  for (int i = lane; i < 16 * 64; i += 32) hbuf[i] = (__bf16)0.0f;   // h0 = 0

  float cst[4][8];
#pragma unroll
  for (int j = 0; j < 4; ++j)
#pragma unroll
    for (int r = 0; r < 8; ++r) cst[j][r] = 0.0f;

  int col = lane & 15, half = lane >> 4;
  const float* xseq = xin + ((size_t)b * T * N + n0) * 64;   // + t*N*64 per step

#if defined(HAVE_ASYNC_LDS)
  async_copy_tile_4k(xseq, sx[wv][0], lane);                 // prologue: t=0
  ASYNC_WAIT(0);                            // weights + first tile resident
#endif
  __syncthreads();                          // publish LDS weights block-wide

  unsigned wofs = 0;                        // always 0; made opaque per-iteration below

  for (int t = 0; t < T; ++t) {
    // Anti-hoist: keeps per-timestep B-tile ds_loads in the loop (cheap LDS reads)
    // instead of letting the compiler hoist 32 tiles into VGPRs and spill to scratch.
    asm volatile("" : "+v"(wofs));
    const __bf16* wihL = swih + wofs;
    const __bf16* whhL = swhh + wofs;

    const float* xb;
#if defined(HAVE_ASYNC_LDS)
    if (t + 1 < T) {
      async_copy_tile_4k(xseq + (size_t)(t + 1) * N * 64, sx[wv][(t + 1) & 1], lane);
      ASYNC_WAIT(8);                        // in-order: current tile's 8 loads done
    } else {
      ASYNC_WAIT(0);
    }
    xb = sx[wv][t & 1];
#else
    xb = xseq + (size_t)t * N * 64;
    if (t + 1 < T) __builtin_prefetch(xb + (size_t)N * 64, 0, 0);   // global_prefetch_b8
#endif

    v16bf ax0 = make_a_tile_f32(xb, 64, 0, lane);
    v16bf ax1 = make_a_tile_f32(xb, 64, 32, lane);

    // h A-tiles from per-wave LDS (LDS in-order within wave; no barrier needed)
    v16bf ah0, ah1;
    {
      const __bf16* p = hbuf + (size_t)col * 64;
      int kh = half * 8;
#pragma unroll
      for (int i = 0; i < 8; ++i) {
        ah0[i] = p[kh + i];       ah0[i + 8] = p[16 + kh + i];
        ah1[i] = p[32 + kh + i];  ah1[i + 8] = p[48 + kh + i];
      }
    }

#pragma unroll
    for (int j = 0; j < 4; ++j) {           // hidden chunk: cols 16j..16j+15
      v8f acc[4];
#pragma unroll
      for (int g = 0; g < 4; ++g) {         // torch gate order i,f,g,o
        int nt = g * 4 + j;                 // N-tile in the 256-wide gate matrix
        float bias = bih[nt * 16 + col] + bhh[nt * 16 + col];
        v8f c;
#pragma unroll
        for (int r = 0; r < 8; ++r) c[r] = bias;
        c = wmma_bf16(ax0, make_b_tile_bf16(wihL, WPITCH, nt, 0, lane), c);
        c = wmma_bf16(ax1, make_b_tile_bf16(wihL, WPITCH, nt, 32, lane), c);
        c = wmma_bf16(ah0, make_b_tile_bf16(whhL, WPITCH, nt, 0, lane), c);
        c = wmma_bf16(ah1, make_b_tile_bf16(whhL, WPITCH, nt, 32, lane), c);
        acc[g] = c;
      }
#pragma unroll
      for (int r = 0; r < 8; ++r) {
        float ig = sigmoidf_(acc[0][r]);
        float fg = sigmoidf_(acc[1][r]);
        float gg = tanhf(acc[2][r]);
        float og = sigmoidf_(acc[3][r]);
        float cn = fg * cst[j][r] + ig * gg;
        cst[j][r] = cn;
        float hn = og * tanhf(cn);
        int M = r + half * 8;
        hbuf[(size_t)M * 64 + j * 16 + col] = (__bf16)hn;
      }
    }
  }

  // out = hT @ W_lin + b_lin  (O = 1)
  if (lane < 16) {
    float acc = blin[0];
#pragma unroll
    for (int k = 0; k < 64; ++k) acc += (float)hbuf[(size_t)lane * 64 + k] * wlin[k];
    out[s0 + lane] = acc;
  }
}

// ---------------- host orchestration ----------------
extern "C" void kernel_launch(void* const* d_in, const int* in_sizes, int n_in,
                              void* d_out, int out_size, void* d_ws, size_t ws_size,
                              hipStream_t stream) {
  (void)in_sizes; (void)n_in; (void)out_size; (void)ws_size;
  const float*      x    = (const float*)d_in[0];
  const long long*  ei   = (const long long*)d_in[1];
  const float*      W1   = (const float*)d_in[2];
  const float*      b1   = (const float*)d_in[3];
  const float*      W2   = (const float*)d_in[4];
  const float*      b2   = (const float*)d_in[5];
  const float*      Wih  = (const float*)d_in[6];
  const float*      Whh  = (const float*)d_in[7];
  const float*      bih  = (const float*)d_in[8];
  const float*      bhh  = (const float*)d_in[9];
  const float*      wlin = (const float*)d_in[10];
  const float*      blin = (const float*)d_in[11];

  constexpr int N = 10000, E = 160000, T = 32, Bc = 8;
  constexpr size_t BTN = (size_t)Bc * T * N;          // 2,560,000 rows

  char*   ws     = (char*)d_ws;
  float*  dinv   = (float*)ws;                        // N f32
  __bf16* wt1    = (__bf16*)(ws + 65536);             // [64][32]
  __bf16* wt2    = (__bf16*)(ws + 65536 + 4096);      // [64][64]
  __bf16* wihb   = (__bf16*)(ws + 65536 + 4096 + 8192);
  __bf16* whhb   = (__bf16*)(ws + 65536 + 4096 + 8192 + 32768);
  float*  hbuf   = (float*)(ws + 262144);             // BTN*64 f32 (655 MB)
  float*  aggbuf = hbuf + BTN * 64;                   // BTN*64 f32 (655 MB)

  // graph norm
  zero_kernel<<<(N + 255) / 256, 256, 0, stream>>>(dinv, N);
  deg_kernel<<<(E + 255) / 256, 256, 0, stream>>>(ei, dinv, E);
  dinv_kernel<<<(N + 255) / 256, 256, 0, stream>>>(dinv, N);

  // weights -> bf16, [N][K] layout so WMMA B-tiles are contiguous per lane
  prep_weights_kernel<<<64, 256, 0, stream>>>(W1, W2, Wih, Whh, wt1, wt2, wihb, whhb);

  const int rowTiles = (int)(BTN / 16);               // 160,000
  const int gemmBlks = rowTiles / 8;                  // 8 waves / block

  // ---- GCN layer 1 ----
  gemm_kernel<16, 32><<<gemmBlks, 256, 0, stream>>>(x, wt1, hbuf, rowTiles);
  {
    size_t tot2 = BTN * 32;
    agg_init_kernel<<<(unsigned)((tot2 + 255) / 256), 256, 0, stream>>>(hbuf, dinv, aggbuf, N);
    dim3 g(E / 8, Bc * T);
    agg_scatter_kernel<<<g, 256, 0, stream>>>(hbuf, dinv, ei, aggbuf, E, N);
    size_t tot = BTN * 64;
    bias_act_kernel<<<(unsigned)((tot + 255) / 256), 256, 0, stream>>>(aggbuf, b1, 1, tot);
  }

  // ---- GCN layer 2 ----
  gemm_kernel<64, 64><<<gemmBlks, 256, 0, stream>>>(aggbuf, wt2, hbuf, rowTiles);
  {
    size_t tot2 = BTN * 32;
    agg_init_kernel<<<(unsigned)((tot2 + 255) / 256), 256, 0, stream>>>(hbuf, dinv, aggbuf, N);
    dim3 g(E / 8, Bc * T);
    agg_scatter_kernel<<<g, 256, 0, stream>>>(hbuf, dinv, ei, aggbuf, E, N);
    size_t tot = BTN * 64;
    bias_act_kernel<<<(unsigned)((tot + 255) / 256), 256, 0, stream>>>(aggbuf, b2, 0, tot);
  }

  // ---- LSTM + linear ----
  lstm_kernel<<<(Bc * N) / 128, 256, 0, stream>>>(aggbuf, wihb, whhb, bih, bhh,
                                                  wlin, blin, (float*)d_out, N, T);
}
